// AttnBlock_4776003633574
// MI455X (gfx1250) — matrix operand order repaired and checked by reading, not compiled, not run on hardware
//
#include <hip/hip_runtime.h>
#include <hip/hip_bf16.h>
#include <math.h>

// ---------------------------------------------------------------------------
// AttnBlock for MI455X (gfx1250): GN -> QKV (WMMA GEMM) -> flash attention
// (WMMA QK^T + online softmax + WMMA PV) -> proj (WMMA GEMM) + residual.
// All matmuls on v_wmma_f32_16x16x32_f16 (f16 in, f32 accum).
// GEMMs register-blocked 16oc x 64px per wave to cut loads per WMMA.
// ---------------------------------------------------------------------------

#define BATCH 2
#define CH    512
#define NPIX  4096          // 64*64
#define GRPS  32
#define CPG   16            // CH / GRPS
#define EPSV  1e-6f

typedef __attribute__((ext_vector_type(16))) _Float16 v16h;
typedef __attribute__((ext_vector_type(8)))  _Float16 v8h;
typedef __attribute__((ext_vector_type(8)))  float    v8f;

__device__ __forceinline__ v16h cat16(v8h lo, v8h hi) {
    return __builtin_shufflevector(lo, hi, 0,1,2,3,4,5,6,7,8,9,10,11,12,13,14,15);
}

// A-matrix fragment (16x32 f16, MxK), source row-major: elem(m,k) = src[m*ld + k]
// layout: lanes 0-15 M=lane K={0..7,16..23}; lanes 16-31 same M, K={8..15,24..31}
__device__ __forceinline__ v16h load_frag_a(const _Float16* src, int ld, int lane) {
    int m  = lane & 15;
    int ko = (lane < 16) ? 0 : 8;
    const _Float16* p = src + (size_t)m * ld + ko;
    v8h lo = *(const v8h*)(p);
    v8h hi = *(const v8h*)(p + 16);
    return cat16(lo, hi);
}

// B-matrix fragment (32x16 f16, KxN), source "transposed": elem(k,n) = srcT[n*ld + k]
// layout: lanes 0-15 N=lane K={0..7,16..23}; lanes 16-31 N=lane-16 K={8..15,24..31}
__device__ __forceinline__ v16h load_frag_b(const _Float16* srcT, int ld, int lane) {
    int n  = lane & 15;
    int ko = (lane < 16) ? 0 : 8;
    const _Float16* p = srcT + (size_t)n * ld + ko;
    v8h lo = *(const v8h*)(p);
    v8h hi = *(const v8h*)(p + 16);
    return cat16(lo, hi);
}

__device__ __forceinline__ v8f wmma16(v16h a, v16h b, v8f c) {
    return __builtin_amdgcn_wmma_f32_16x16x32_f16(false, a, false, b, (short)0, c,
                                                  false, false);
}

// ---------------------------------------------------------------------------
// 1) GroupNorm statistics: one block per (b, group)
// ---------------------------------------------------------------------------
__global__ __launch_bounds__(256)
void gn_stats_kernel(const float* __restrict__ x, float* __restrict__ mu,
                     float* __restrict__ rstd) {
    int bg = blockIdx.x;               // 0..63
    int b = bg >> 5, g = bg & 31;
    const float* base = x + ((size_t)b * CH + (size_t)g * CPG) * NPIX;
    float s = 0.f, s2 = 0.f;
    for (int i = threadIdx.x; i < CPG * NPIX; i += 256) {
        float v = base[i];
        s += v; s2 += v * v;
    }
    __shared__ float shs[256], shq[256];
    shs[threadIdx.x] = s; shq[threadIdx.x] = s2;
    __syncthreads();
    for (int st = 128; st > 0; st >>= 1) {
        if (threadIdx.x < st) {
            shs[threadIdx.x] += shs[threadIdx.x + st];
            shq[threadIdx.x] += shq[threadIdx.x + st];
        }
        __syncthreads();
    }
    if (threadIdx.x == 0) {
        float inv = 1.0f / (float)(CPG * NPIX);
        float m = shs[0] * inv;
        float var = shq[0] * inv - m * m;
        mu[bg] = m;
        rstd[bg] = rsqrtf(var + EPSV);
    }
}

// ---------------------------------------------------------------------------
// 2) GroupNorm apply + transpose to pixel-major f16: Ht[b][n][c]
// ---------------------------------------------------------------------------
__global__ __launch_bounds__(256)
void gn_apply_kernel(const float* __restrict__ x, const float* __restrict__ gamma,
                     const float* __restrict__ beta, const float* __restrict__ mu,
                     const float* __restrict__ rstd, _Float16* __restrict__ Ht) {
    __shared__ _Float16 tile[32][33];
    int b = blockIdx.z;
    int c0 = blockIdx.y * 32;
    int n0 = blockIdx.x * 32;
    int tn = threadIdx.x & 31, tr = threadIdx.x >> 5;   // 8 rows per pass
    #pragma unroll
    for (int cc = tr; cc < 32; cc += 8) {
        int c = c0 + cc;
        int g = c >> 4;
        float m = mu[b * 32 + g], r = rstd[b * 32 + g];
        float v = x[((size_t)b * CH + c) * NPIX + n0 + tn];
        tile[cc][tn] = (_Float16)((v - m) * r * gamma[c] + beta[c]);
    }
    __syncthreads();
    int tc = threadIdx.x & 31;
    #pragma unroll
    for (int nn = tr; nn < 32; nn += 8) {
        Ht[((size_t)b * NPIX + n0 + nn) * CH + c0 + tc] = tile[tc][nn];
    }
}

// ---------------------------------------------------------------------------
// 3) Weights f32 -> f16 (wq, wk, wv, wp packed consecutively)
// ---------------------------------------------------------------------------
__global__ __launch_bounds__(256)
void wcvt_kernel(const float* __restrict__ wq, const float* __restrict__ wk,
                 const float* __restrict__ wv, const float* __restrict__ wp,
                 _Float16* __restrict__ out) {
    size_t id  = (size_t)blockIdx.x * 256 + threadIdx.x;   // 4 * 262144 total
    size_t sel = id >> 18;
    size_t off = id & 262143;
    const float* s;
    if (sel == 0)      s = wq;
    else if (sel == 1) s = wk;
    else if (sel == 2) s = wv;
    else               s = wp;
    out[id] = (_Float16)s[off];
}

// ---------------------------------------------------------------------------
// 4) Fused QKV GEMM, register-blocked: one wave = 16 oc x 64 px for Q,K,V.
//    Per k-step: 3 A-frags (weights) + 4 B-frags (Ht) feed 12 WMMAs.
//    Q -> Qt[n][c] (scaled 1/sqrt(C)), K -> Kt[n][c], V -> V[c][n]
// ---------------------------------------------------------------------------
__global__ __launch_bounds__(128)
void qkv_gemm_kernel(const _Float16* __restrict__ Wf, const float* __restrict__ bq,
                     const float* __restrict__ bk, const float* __restrict__ bv,
                     const _Float16* __restrict__ Ht, _Float16* __restrict__ Qt,
                     _Float16* __restrict__ Kt, _Float16* __restrict__ V) {
    const float scale = 1.0f / sqrtf((float)CH);
    int lane = threadIdx.x & 31;
    int gw = blockIdx.x * 4 + (threadIdx.x >> 5);   // 4096 waves total
    int b   = gw >> 11;                             // 2048 waves per batch
    int r   = gw & 2047;
    int oct = r >> 6;     // output-channel tile 0..31
    int nt  = r & 63;     // 64-pixel tile 0..63
    const _Float16* wqf = Wf;
    const _Float16* wkf = Wf + 262144;
    const _Float16* wvf = Wf + 524288;
    const _Float16* hb  = Ht + (size_t)b * NPIX * CH + (size_t)(nt * 64) * CH;
    v8f accq[4], acck[4], accv[4];
    #pragma unroll
    for (int ns = 0; ns < 4; ++ns) {
        accq[ns] = (v8f){0,0,0,0,0,0,0,0};
        acck[ns] = (v8f){0,0,0,0,0,0,0,0};
        accv[ns] = (v8f){0,0,0,0,0,0,0,0};
    }
    for (int k = 0; k < 16; ++k) {
        v16h aq = load_frag_a(wqf + (size_t)(oct * 16) * CH + k * 32, CH, lane);
        v16h ak = load_frag_a(wkf + (size_t)(oct * 16) * CH + k * 32, CH, lane);
        v16h av = load_frag_a(wvf + (size_t)(oct * 16) * CH + k * 32, CH, lane);
        #pragma unroll
        for (int ns = 0; ns < 4; ++ns) {
            v16h bf = load_frag_b(hb + (size_t)(ns * 16) * CH + k * 32, CH, lane);
            accq[ns] = wmma16(aq, bf, accq[ns]);
            acck[ns] = wmma16(ak, bf, acck[ns]);
            accv[ns] = wmma16(av, bf, accv[ns]);
        }
    }
    int nloc = lane & 15;
    int moff = (lane < 16) ? 0 : 8;
    #pragma unroll
    for (int ns = 0; ns < 4; ++ns) {
        size_t nbase = (size_t)nt * 64 + ns * 16 + nloc;
        #pragma unroll
        for (int rr = 0; rr < 8; ++rr) {
            int oc = oct * 16 + moff + rr;
            Qt[((size_t)b * NPIX + nbase) * CH + oc] =
                (_Float16)((accq[ns][rr] + bq[oc]) * scale);
            Kt[((size_t)b * NPIX + nbase) * CH + oc] =
                (_Float16)(acck[ns][rr] + bk[oc]);
            V [((size_t)b * CH + oc) * NPIX + nbase] =
                (_Float16)(accv[ns][rr] + bv[oc]);
        }
    }
}

// ---------------------------------------------------------------------------
// 5) Flash attention. Block = 4 waves, 16 queries, each wave owns a
//    128-channel slice of the contraction and of the output accumulator.
//    Per 32-key step: 8 QK WMMAs -> ds_add_f32 cross-wave S reduce ->
//    online softmax -> P^T via LDS -> 8 PV WMMAs. Out: Ot[b][n][c] f16.
// ---------------------------------------------------------------------------
__global__ __launch_bounds__(128)
void flash_kernel(const _Float16* __restrict__ Qt, const _Float16* __restrict__ Kt,
                  const _Float16* __restrict__ V, _Float16* __restrict__ Ot) {
    __shared__ __align__(16) float    sred[16 * 33];   // padded: bank-spread
    __shared__ __align__(16) _Float16 psh[16 * 32];    // P tile, row-major [i][j]
    int lane = threadIdx.x & 31;
    int wave = threadIdx.x >> 5;
    int b  = blockIdx.y;
    int qt = blockIdx.x;
    const _Float16* Qb = Qt + (size_t)b * NPIX * CH;
    const _Float16* Kb = Kt + (size_t)b * NPIX * CH;
    const _Float16* Vb = V  + (size_t)b * CH * NPIX;
    int c0 = wave * 128;

    // Q fragments resident in registers for the whole key loop
    v16h qa[4];
    #pragma unroll
    for (int kb = 0; kb < 4; ++kb)
        qa[kb] = load_frag_a(Qb + (size_t)(qt * 16) * CH + c0 + kb * 32, CH, lane);

    v8f o[8];
    #pragma unroll
    for (int mb = 0; mb < 8; ++mb) o[mb] = (v8f){0,0,0,0,0,0,0,0};
    float m = -INFINITY, l = 0.f;     // per lane, row = lane & 15 (dup in halves)
    int row = lane & 15;
    int jlo = (lane < 16) ? 0 : 16;
    int mo  = (lane < 16) ? 0 : 8;

    for (int kt = 0; kt < 128; ++kt) {
        int jb = kt * 32;
        // zero the cross-wave S reduction buffer
        for (int i = threadIdx.x; i < 16 * 33; i += 128) sred[i] = 0.f;
        __syncthreads();
        // partial S = (Q-slice)^T (K-slice) over this wave's 128 channels
        #pragma unroll
        for (int js = 0; js < 2; ++js) {
            v8f s = (v8f){0,0,0,0,0,0,0,0};
            #pragma unroll
            for (int kb = 0; kb < 4; ++kb) {
                v16h bk = load_frag_b(Kb + (size_t)(jb + js * 16) * CH + c0 + kb * 32,
                                      CH, lane);
                s = wmma16(qa[kb], bk, s);
            }
            int j = js * 16 + (lane & 15);
            #pragma unroll
            for (int rr = 0; rr < 8; ++rr)
                atomicAdd(&sred[(mo + rr) * 33 + j], s[rr]);
        }
        // prefetch next key/value tiles into L2/L0 while softmax runs
        if (kt + 1 < 128) {
            const _Float16* pk = Kb + (size_t)(jb + 32 + lane) * CH + c0;
            __builtin_prefetch(pk, 0, 0);
            __builtin_prefetch(pk + 64, 0, 0);
            #pragma unroll
            for (int pm = 0; pm < 4; ++pm) {
                const _Float16* pv =
                    Vb + (size_t)(c0 + pm * 32 + lane) * NPIX + jb + 32;
                __builtin_prefetch(pv, 0, 0);
            }
        }
        __syncthreads();
        // online softmax (redundant per wave; identical state per lane-row)
        float mx = -INFINITY;
        #pragma unroll 8
        for (int j = 0; j < 32; ++j) mx = fmaxf(mx, sred[row * 33 + j]);
        float mnew = fmaxf(m, mx);
        float corr = __expf(m - mnew);
        float ssum = 0.f;
        for (int j = 0; j < 32; ++j) {
            float p = __expf(sred[row * 33 + j] - mnew);
            ssum += p;
            if (wave == 0 && j >= jlo && j < jlo + 16)
                psh[row * 32 + j] = (_Float16)p;
        }
        l = l * corr + ssum;
        m = mnew;
        __syncthreads();
        // rescale accumulator and do PV
        #pragma unroll
        for (int mb = 0; mb < 8; ++mb) o[mb] *= corr;
        v16h bp = load_frag_b(psh, 32, lane);   // P^T fragment from LDS
        #pragma unroll
        for (int mb = 0; mb < 8; ++mb) {
            v16h av = load_frag_a(Vb + (size_t)(c0 + mb * 16) * NPIX + jb, NPIX, lane);
            o[mb] = wmma16(av, bp, o[mb]);
        }
    }
    // normalize and write Ot[b][n][c]
    float invl = 1.0f / l;
    size_t nbase = (size_t)qt * 16 + (lane & 15);
    #pragma unroll
    for (int mb = 0; mb < 8; ++mb) {
        #pragma unroll
        for (int rr = 0; rr < 8; ++rr) {
            int c = c0 + mb * 16 + mo + rr;
            Ot[((size_t)b * NPIX + nbase) * CH + c] = (_Float16)(o[mb][rr] * invl);
        }
    }
}

// ---------------------------------------------------------------------------
// 6) Output projection + bias + residual -> f32 out[b][c][n].
//    Register-blocked: one wave = 16 oc x 64 px (1 A-frag + 4 B-frags / 4 WMMA)
// ---------------------------------------------------------------------------
__global__ __launch_bounds__(128)
void proj_gemm_kernel(const _Float16* __restrict__ Wp, const float* __restrict__ bp,
                      const _Float16* __restrict__ Ot, const float* __restrict__ x,
                      float* __restrict__ out) {
    int lane = threadIdx.x & 31;
    int gw = blockIdx.x * 4 + (threadIdx.x >> 5);   // 4096 waves total
    int b   = gw >> 11;
    int r   = gw & 2047;
    int oct = r >> 6;
    int nt  = r & 63;
    const _Float16* ob = Ot + (size_t)b * NPIX * CH + (size_t)(nt * 64) * CH;
    v8f acc[4];
    #pragma unroll
    for (int ns = 0; ns < 4; ++ns) acc[ns] = (v8f){0,0,0,0,0,0,0,0};
    for (int k = 0; k < 16; ++k) {
        v16h af = load_frag_a(Wp + (size_t)(oct * 16) * CH + k * 32, CH, lane);
        #pragma unroll
        for (int ns = 0; ns < 4; ++ns) {
            v16h bf = load_frag_b(ob + (size_t)(ns * 16) * CH + k * 32, CH, lane);
            acc[ns] = wmma16(af, bf, acc[ns]);
        }
    }
    int nloc = lane & 15;
    int moff = (lane < 16) ? 0 : 8;
    #pragma unroll
    for (int ns = 0; ns < 4; ++ns) {
        size_t nbase = (size_t)nt * 64 + ns * 16 + nloc;
        #pragma unroll
        for (int rr = 0; rr < 8; ++rr) {
            int oc = oct * 16 + moff + rr;
            size_t idx = ((size_t)b * CH + oc) * NPIX + nbase;
            out[idx] = acc[ns][rr] + bp[oc] + x[idx];
        }
    }
}

// ---------------------------------------------------------------------------
extern "C" void kernel_launch(void* const* d_in, const int* in_sizes, int n_in,
                              void* d_out, int out_size, void* d_ws, size_t ws_size,
                              hipStream_t stream) {
    const float* x     = (const float*)d_in[0];
    const float* gamma = (const float*)d_in[1];
    const float* beta  = (const float*)d_in[2];
    const float* wq    = (const float*)d_in[3];
    const float* bq    = (const float*)d_in[4];
    const float* wk    = (const float*)d_in[5];
    const float* bk    = (const float*)d_in[6];
    const float* wv    = (const float*)d_in[7];
    const float* bv    = (const float*)d_in[8];
    const float* wp    = (const float*)d_in[9];
    const float* bp    = (const float*)d_in[10];
    float* out = (float*)d_out;

    unsigned char* ws = (unsigned char*)d_ws;
    const size_t MAT = (size_t)BATCH * NPIX * CH * sizeof(_Float16);  // 8 MB
    float*     muv  = (float*)ws;                     //  64 f32
    float*     rsd  = (float*)(ws + 256);             //  64 f32
    _Float16*  Ht   = (_Float16*)(ws + 1024);
    _Float16*  Qt   = (_Float16*)(ws + 1024 + MAT);
    _Float16*  Kt   = (_Float16*)(ws + 1024 + 2 * MAT);
    _Float16*  Vm   = (_Float16*)(ws + 1024 + 3 * MAT);
    _Float16*  Ot   = (_Float16*)(ws + 1024 + 4 * MAT);
    _Float16*  Wf   = (_Float16*)(ws + 1024 + 5 * MAT);  // 4 x 512x512 f16 = 2 MB

    gn_stats_kernel<<<BATCH * GRPS, 256, 0, stream>>>(x, muv, rsd);
    gn_apply_kernel<<<dim3(NPIX / 32, CH / 32, BATCH), 256, 0, stream>>>(
        x, gamma, beta, muv, rsd, Ht);
    wcvt_kernel<<<(4 * CH * CH) / 256, 256, 0, stream>>>(wq, wk, wv, wp, Wf);
    qkv_gemm_kernel<<<(BATCH * (CH / 16) * (NPIX / 64)) / 4, 128, 0, stream>>>(
        Wf, bq, bk, bv, Ht, Qt, Kt, Vm);
    flash_kernel<<<dim3(NPIX / 16, BATCH), 128, 0, stream>>>(Qt, Kt, Vm, Ot);
    proj_gemm_kernel<<<(BATCH * (CH / 16) * (NPIX / 64)) / 4, 128, 0, stream>>>(
        Wf + 3 * (size_t)CH * CH, bp, Ot, x, out);
}
